// Decoder_72516227825826
// MI455X (gfx1250) — matrix operand order repaired and checked by reading, not compiled, not run on hardware
//
#include <hip/hip_runtime.h>
#include <hip/hip_bf16.h>
#include <stdint.h>

// ---------------- problem dimensions ----------------
#define DB   64          // batch
#define DP   196         // pixels (attention positions)
#define DENC 2048        // encoder dim
#define DH   1024        // hidden
#define DA   1024        // attention dim
#define DE   512         // embedding dim
#define DV   10000       // vocab
#define DTC  32          // caption length
#define DT   31          // decode steps = DTC-1
#define DX   (DE + DENC) // 2560 (LSTM input)
#define DG   (4 * DH)    // 4096 (gates)
#define MEO  (DB * DP)   // 12544 rows of enc_out

typedef __attribute__((ext_vector_type(16))) __bf16 v16bf;
typedef __attribute__((ext_vector_type(2)))  __bf16 v2bf;
typedef __attribute__((ext_vector_type(8)))  float  v8f;
typedef unsigned short u16;
typedef unsigned int   u32;

// ---------------- bf16 helpers ----------------
// Packed f32x2 -> bf16x2 (RNE, branchless; finite inputs by construction).
// Uses V_PERM_B32 to gather the two rounded high halves in one instruction.
__device__ __forceinline__ u32 pack_bf16x2(float a, float b) {
#if __has_builtin(__builtin_amdgcn_cvt_pk_bf16_f32)
    const v2bf r = __builtin_amdgcn_cvt_pk_bf16_f32(a, b);
    union { v2bf v; u32 u; } x; x.v = r; return x.u;
#else
    const u32 ua = __float_as_uint(a), ub = __float_as_uint(b);
    const u32 ra = ua + 0x7fffu + ((ua >> 16) & 1u);
    const u32 rb = ub + 0x7fffu + ((ub >> 16) & 1u);
#if __has_builtin(__builtin_amdgcn_perm)
    // dst bytes: [0]=ra.b2 [1]=ra.b3 [2]=rb.b2 [3]=rb.b3  (sel 0-3 -> SRC1, 4-7 -> SRC0)
    return __builtin_amdgcn_perm(rb, ra, 0x07060302u);
#else
    return (ra >> 16) | (rb & 0xffff0000u);
#endif
#endif
}
__device__ __forceinline__ __bf16 bits_bf16(u16 s) {
    union { u16 u; __bf16 b; } x; x.u = s; return x.b;
}
__device__ __forceinline__ float sigmoidf(float x) { return 1.0f / (1.0f + expf(-x)); }

// ---------------- f32 -> bf16 weight conversion (4 elems/thread, packed b64 store) ----------------
__global__ void k_conv_bf16(const float* __restrict__ src, u32* __restrict__ dst, int n4) {
    const int i = blockIdx.x * 256 + threadIdx.x;
    if (i < n4) {
        const float4 f = reinterpret_cast<const float4*>(src)[i];
        uint2 r;
        r.x = pack_bf16x2(f.x, f.y);
        r.y = pack_bf16x2(f.z, f.w);
        reinterpret_cast<uint2*>(dst)[i] = r;
    }
}

// ---------------- prep: stable argsort desc, gather caps, row map ----------------
__global__ __launch_bounds__(256)
void k_prep(const int* __restrict__ lengths, const int* __restrict__ caps_in,
            int* __restrict__ ind, int* __restrict__ dlen, int* __restrict__ caps,
            int* __restrict__ rmap, int* __restrict__ out_caps,
            int* __restrict__ out_dlen, int* __restrict__ out_ind) {
    __shared__ int sInd[DB];
    const int tid = threadIdx.x;
    if (tid < DB) {
        const int li = lengths[tid];
        int rank = 0;
        for (int j = 0; j < DB; ++j) {
            const int lj = lengths[j];
            if (lj > li || (lj == li && j < tid)) ++rank;   // stable, descending
        }
        sInd[rank] = tid;
    }
    __syncthreads();
    if (tid < DB) {
        const int src = sInd[tid];
        ind[tid] = src;              out_ind[tid] = src;
        const int dl = lengths[src] - 1;
        dlen[tid] = dl;              out_dlen[tid] = dl;
    }
    __syncthreads();
    for (int i = tid; i < DB * DTC; i += 256) {
        const int b = i / DTC, j = i % DTC;
        const int v = caps_in[sInd[b] * DTC + j];
        caps[i] = v; out_caps[i] = v;
    }
    for (int r = tid; r < MEO; r += 256)
        rmap[r] = sInd[r / DP] * DP + (r % DP);
}

// ---------------- mean over P of permuted enc_out ----------------
__global__ void k_mean(const float* __restrict__ enc_out, const int* __restrict__ ind,
                       float* __restrict__ mean) {
    const int e = blockIdx.x * 256 + threadIdx.x;
    const int b = blockIdx.y;
    const float* base = enc_out + ((size_t)ind[b] * DP) * DENC + e;
    float s = 0.f;
    for (int p = 0; p < DP; ++p) s += base[(size_t)p * DENC];
    mean[b * DENC + e] = s * (1.0f / (float)DP);
}

// ---------------- bf16 WMMA skinny GEMM:  C[M,N] (+)= A[M,K] * W[N,K]^T + bias ----------------
// Block: 256 threads = 8 waves covering 64(M) x 64(N); wave w: m-tile = w&3, n-group = w>>2.
// A (f32) staged per 32-K chunk through double-buffered LDS as packed bf16 pairs;
// W pre-converted bf16 row-major (N,K). LDS u32 stride 17 (=34 bf16) -> conflict-free.
__global__ __launch_bounds__(256)
void k_gemm_bf16(const float* __restrict__ Af, int M, int K,
                 const u16* __restrict__ Wb, int N,
                 const float* __restrict__ bias,
                 float* __restrict__ C, int accumulate,
                 const int* __restrict__ row_map) {
    __shared__ u32 lsA[2][64 * 17];
    const int tid   = threadIdx.x;
    const int lane  = tid & 31;
    const int w     = tid >> 5;
    const int mt    = w & 3;                      // m-tile 0..3
    const int khalf = lane >> 4;
    const int l15   = lane & 15;
    const int m0    = blockIdx.y * 64;
    const int nbase = blockIdx.x * 64 + (w >> 2) * 32;

    v8f acc0 = {}; v8f acc1 = {};

    // cooperative A-chunk loader: thread -> (row, 8 cols)
    const int lrow = tid >> 2;                    // 0..63
    const int lcol = (tid & 3) * 8;               // 0,8,16,24
    int srow = m0 + lrow;
    if (row_map) srow = row_map[srow];
    const float* aRow = Af + (size_t)srow * K;

    // W row pointers for the two n-tiles this wave owns
    int n0 = nbase + l15;      if (n0 >= N) n0 = N - 1;
    int n1 = nbase + 16 + l15; if (n1 >= N) n1 = N - 1;
    const u16* w0 = Wb + (size_t)n0 * K + 16 * khalf;
    const u16* w1 = Wb + (size_t)n1 * K + 16 * khalf;

    const int nChunks = K >> 5;

    // preload chunk 0
    float4 f0 = *reinterpret_cast<const float4*>(aRow + lcol);
    float4 f1 = *reinterpret_cast<const float4*>(aRow + lcol + 4);
    {
        u32* d = &lsA[0][lrow * 17 + (lcol >> 1)];
        d[0] = pack_bf16x2(f0.x, f0.y); d[1] = pack_bf16x2(f0.z, f0.w);
        d[2] = pack_bf16x2(f1.x, f1.y); d[3] = pack_bf16x2(f1.z, f1.w);
    }

    int buf = 0;
    const int am = mt * 16 + l15;
    for (int ci = 0; ci < nChunks; ++ci) {
        __syncthreads();
        const int k0 = ci << 5;
        const bool hasNext = (ci + 1 < nChunks);
        if (hasNext) {
            f0 = *reinterpret_cast<const float4*>(aRow + k0 + 32 + lcol);
            f1 = *reinterpret_cast<const float4*>(aRow + k0 + 32 + lcol + 4);
        }

        // A fragment (16x32 bf16): lane L<16 -> row L, K 0..15; L>=16 same row, K 16..31 (contiguous)
        v16bf afrag;
        #pragma unroll
        for (int v = 0; v < 8; ++v) {
            const int k = 16 * khalf + 2 * v;
            const u32 pair = lsA[buf][am * 17 + (k >> 1)];
            afrag[2 * v]     = bits_bf16((u16)(pair & 0xffffu));
            afrag[2 * v + 1] = bits_bf16((u16)(pair >> 16));
        }

        // B fragments: lane = col n; lanes 0-15 hold K 0..15, lanes 16-31 hold K 16..31 (contiguous)
        {
            const v16bf bfrag = *reinterpret_cast<const v16bf*>(w0 + k0);
            acc0 = __builtin_amdgcn_wmma_f32_16x16x32_bf16(false, afrag, false, bfrag,
                                                           (short)0, acc0, false, false);
        }
        {
            const v16bf bfrag = *reinterpret_cast<const v16bf*>(w1 + k0);
            acc1 = __builtin_amdgcn_wmma_f32_16x16x32_bf16(false, afrag, false, bfrag,
                                                           (short)0, acc1, false, false);
        }

        if (hasNext) {
            u32* d = &lsA[buf ^ 1][lrow * 17 + (lcol >> 1)];
            d[0] = pack_bf16x2(f0.x, f0.y); d[1] = pack_bf16x2(f0.z, f0.w);
            d[2] = pack_bf16x2(f1.x, f1.y); d[3] = pack_bf16x2(f1.z, f1.w);
        }
        buf ^= 1;
    }

    // D layout: m = r + 8*(lane>>4), n = lane&15
    #pragma unroll
    for (int r = 0; r < 8; ++r) {
        const int m = m0 + mt * 16 + 8 * khalf + r;
        int n = nbase + l15;
        if (n < N) {
            float v = acc0[r] + (bias ? bias[n] : 0.0f);
            if (accumulate) v += C[(size_t)m * N + n];
            C[(size_t)m * N + n] = v;
        }
        n = nbase + 16 + l15;
        if (n < N) {
            float v = acc1[r] + (bias ? bias[n] : 0.0f);
            if (accumulate) v += C[(size_t)m * N + n];
            C[(size_t)m * N + n] = v;
        }
    }
}

// ---------------- attention scores: relu(enc_att + dec) . W_full + b_full ----------------
__global__ __launch_bounds__(256)
void k_attn_scores(const float* __restrict__ encAtt, const float* __restrict__ dec,
                   const float* __restrict__ Wfull, const float* __restrict__ bfull,
                   float* __restrict__ scores) {
    const int row  = blockIdx.x * 8 + (threadIdx.x >> 5);  // one wave per (b,p) row
    const int lane = threadIdx.x & 31;
    const int b    = row / DP;
    const float* ea = encAtt + (size_t)row * DA;
    const float* dc = dec + (size_t)b * DA;
    float s = 0.f;
    for (int k = lane; k < DA; k += 32) {
        float v = ea[k] + dc[k];
        v = v > 0.f ? v : 0.f;
        s = fmaf(v, Wfull[k], s);
    }
    #pragma unroll
    for (int off = 16; off > 0; off >>= 1) s += __shfl_down(s, off, 32);
    if (lane == 0) scores[row] = s + bfull[0];
}

// ---------------- softmax over P; emit masked alphas to output ----------------
__global__ __launch_bounds__(256)
void k_softmax(const float* __restrict__ scores, float* __restrict__ alpha,
               float* __restrict__ outAlphas, const int* __restrict__ dlen, int t) {
    __shared__ float red[256];
    const int b = blockIdx.x, tid = threadIdx.x;
    float v = (tid < DP) ? scores[b * DP + tid] : -3.0e38f;
    red[tid] = v; __syncthreads();
    for (int s = 128; s > 0; s >>= 1) { if (tid < s) red[tid] = fmaxf(red[tid], red[tid + s]); __syncthreads(); }
    const float mx = red[0]; __syncthreads();
    float e = (tid < DP) ? expf(v - mx) : 0.f;
    red[tid] = e; __syncthreads();
    for (int s = 128; s > 0; s >>= 1) { if (tid < s) red[tid] += red[tid + s]; __syncthreads(); }
    const float inv = 1.0f / red[0];
    if (tid < DP) {
        const float a = e * inv;
        alpha[b * DP + tid] = a;
        const float mf = (t < dlen[b]) ? 1.0f : 0.0f;
        outAlphas[((size_t)b * DT + t) * DP + tid] = a * mf;
    }
}

// ---------------- context = sum_p eo[b,p,:] * alpha[b,p] ----------------
__global__ void k_context(const float* __restrict__ enc_out, const int* __restrict__ ind,
                          const float* __restrict__ alpha, float* __restrict__ ctx) {
    const int e = blockIdx.x * 256 + threadIdx.x;
    const int b = blockIdx.y;
    const float* base = enc_out + ((size_t)ind[b] * DP) * DENC + e;
    const float* al = alpha + b * DP;
    float s = 0.f;
    for (int p = 0; p < DP; ++p) s = fmaf(base[(size_t)p * DENC], al[p], s);
    ctx[b * DENC + e] = s;
}

// ---------------- x = [emb_t | sigmoid(gate)*context] ----------------
__global__ void k_build_x(const float* __restrict__ emb, const int* __restrict__ caps, int t,
                          const float* __restrict__ gate, const float* __restrict__ ctx,
                          float* __restrict__ x) {
    const int idx = blockIdx.x * 256 + threadIdx.x;   // DB*DX
    const int b = idx / DX, j = idx % DX;
    if (j < DE) {
        const int cap = caps[b * DTC + t];
        x[idx] = emb[(size_t)cap * DE + j];
    } else {
        const int e2 = j - DE;
        x[idx] = sigmoidf(gate[b * DENC + e2]) * ctx[b * DENC + e2];
    }
}

// ---------------- LSTM pointwise + masked state update ----------------
__global__ void k_lstm(const float* __restrict__ gates, float* __restrict__ h,
                       float* __restrict__ c, float* __restrict__ hnew,
                       const int* __restrict__ dlen, int t) {
    const int idx = blockIdx.x * 256 + threadIdx.x;   // DB*DH
    const int b = idx / DH, j = idx % DH;
    const float* g = gates + (size_t)b * DG;
    const float ig = sigmoidf(g[j]);
    const float fg = sigmoidf(g[DH + j]);
    const float gg = tanhf(g[2 * DH + j]);
    const float og = sigmoidf(g[3 * DH + j]);
    const float cn = fg * c[idx] + ig * gg;
    const float hn = og * tanhf(cn);
    hnew[idx] = hn;
    if (t < dlen[b]) { h[idx] = hn; c[idx] = cn; }
}

// ---------------- masked prediction write ----------------
__global__ void k_write_preds(const float* __restrict__ preds, const int* __restrict__ dlen,
                              int t, float* __restrict__ out) {
    const int idx = blockIdx.x * 256 + threadIdx.x;   // DB*DV
    const int b = idx / DV, v = idx % DV;
    const float mf = (t < dlen[b]) ? 1.f : 0.f;
    out[((size_t)b * DT + t) * DV + v] = preds[idx] * mf;
}

// ---------------- workspace layout (bytes, ~128 MB total) ----------------
static constexpr size_t al256(size_t x) { return (x + 255) & ~(size_t)255; }
static constexpr size_t OFF_WENC = 0;
static constexpr size_t OFF_WDEC = al256(OFF_WENC + (size_t)DA * DENC * 2);
static constexpr size_t OFF_WBETA= al256(OFF_WDEC + (size_t)DA * DH * 2);
static constexpr size_t OFF_WIH  = al256(OFF_WBETA+ (size_t)DENC * DH * 2);
static constexpr size_t OFF_WHH  = al256(OFF_WIH  + (size_t)DG * DX * 2);
static constexpr size_t OFF_WFC  = al256(OFF_WHH  + (size_t)DG * DH * 2);
static constexpr size_t OFF_WINH = al256(OFF_WFC  + (size_t)DV * DH * 2);
static constexpr size_t OFF_WINC = al256(OFF_WINH + (size_t)DH * DENC * 2);
static constexpr size_t OFF_EATT = al256(OFF_WINC + (size_t)DH * DENC * 2);
static constexpr size_t OFF_MEAN = al256(OFF_EATT + (size_t)MEO * DA * 4);
static constexpr size_t OFF_H    = al256(OFF_MEAN + (size_t)DB * DENC * 4);
static constexpr size_t OFF_C    = al256(OFF_H    + (size_t)DB * DH * 4);
static constexpr size_t OFF_HN   = al256(OFF_C    + (size_t)DB * DH * 4);
static constexpr size_t OFF_DEC  = al256(OFF_HN   + (size_t)DB * DH * 4);
static constexpr size_t OFF_SCR  = al256(OFF_DEC  + (size_t)DB * DA * 4);
static constexpr size_t OFF_ALPH = al256(OFF_SCR  + (size_t)DB * DP * 4);
static constexpr size_t OFF_CTX  = al256(OFF_ALPH + (size_t)DB * DP * 4);
static constexpr size_t OFF_GATE = al256(OFF_CTX  + (size_t)DB * DENC * 4);
static constexpr size_t OFF_X    = al256(OFF_GATE + (size_t)DB * DENC * 4);
static constexpr size_t OFF_GTS  = al256(OFF_X    + (size_t)DB * DX * 4);
static constexpr size_t OFF_PRD  = al256(OFF_GTS  + (size_t)DB * DG * 4);
static constexpr size_t OFF_IND  = al256(OFF_PRD  + (size_t)DB * DV * 4);
static constexpr size_t OFF_DLEN = al256(OFF_IND  + DB * 4);
static constexpr size_t OFF_CAPS = al256(OFF_DLEN + DB * 4);
static constexpr size_t OFF_RMAP = al256(OFF_CAPS + (size_t)DB * DTC * 4);

// output element offsets
static constexpr size_t O_PRED  = 0;
static constexpr size_t O_ALPHA = (size_t)DB * DT * DV;
static constexpr size_t O_CAPS  = O_ALPHA + (size_t)DB * DT * DP;
static constexpr size_t O_DLEN  = O_CAPS + (size_t)DB * DTC;
static constexpr size_t O_IND   = O_DLEN + DB;

extern "C" void kernel_launch(void* const* d_in, const int* in_sizes, int n_in,
                              void* d_out, int out_size, void* d_ws, size_t ws_size,
                              hipStream_t stream) {
    const float* enc_out   = (const float*)d_in[0];
    const int*   caps_in   = (const int*)  d_in[1];
    const int*   lengths   = (const int*)  d_in[2];
    const float* W_enc_att = (const float*)d_in[3];
    const float* b_enc_att = (const float*)d_in[4];
    const float* W_dec_att = (const float*)d_in[5];
    const float* b_dec_att = (const float*)d_in[6];
    const float* W_full    = (const float*)d_in[7];
    const float* b_full    = (const float*)d_in[8];
    const float* emb       = (const float*)d_in[9];
    const float* W_ih      = (const float*)d_in[10];
    const float* b_ih      = (const float*)d_in[11];
    const float* W_hh      = (const float*)d_in[12];
    const float* b_hh      = (const float*)d_in[13];
    const float* W_init_h  = (const float*)d_in[14];
    const float* b_init_h  = (const float*)d_in[15];
    const float* W_init_c  = (const float*)d_in[16];
    const float* b_init_c  = (const float*)d_in[17];
    const float* W_beta    = (const float*)d_in[18];
    const float* b_beta    = (const float*)d_in[19];
    const float* W_fc      = (const float*)d_in[20];
    const float* b_fc      = (const float*)d_in[21];

    char* ws = (char*)d_ws;
    u16* wenc  = (u16*)(ws + OFF_WENC);
    u16* wdec  = (u16*)(ws + OFF_WDEC);
    u16* wbeta = (u16*)(ws + OFF_WBETA);
    u16* wih   = (u16*)(ws + OFF_WIH);
    u16* whh   = (u16*)(ws + OFF_WHH);
    u16* wfc   = (u16*)(ws + OFF_WFC);
    u16* winh  = (u16*)(ws + OFF_WINH);
    u16* winc  = (u16*)(ws + OFF_WINC);
    float* eatt = (float*)(ws + OFF_EATT);
    float* mean = (float*)(ws + OFF_MEAN);
    float* h    = (float*)(ws + OFF_H);
    float* c    = (float*)(ws + OFF_C);
    float* hn   = (float*)(ws + OFF_HN);
    float* dec  = (float*)(ws + OFF_DEC);
    float* scr  = (float*)(ws + OFF_SCR);
    float* alph = (float*)(ws + OFF_ALPH);
    float* ctx  = (float*)(ws + OFF_CTX);
    float* gate = (float*)(ws + OFF_GATE);
    float* x    = (float*)(ws + OFF_X);
    float* gts  = (float*)(ws + OFF_GTS);
    float* prd  = (float*)(ws + OFF_PRD);
    int* ind  = (int*)(ws + OFF_IND);
    int* dlen = (int*)(ws + OFF_DLEN);
    int* caps = (int*)(ws + OFF_CAPS);
    int* rmap = (int*)(ws + OFF_RMAP);

    float* outf = (float*)d_out;
    float* outPred  = outf + O_PRED;
    float* outAlpha = outf + O_ALPHA;
    int*   outCaps  = (int*)(outf + O_CAPS);
    int*   outDlen  = (int*)(outf + O_DLEN);
    int*   outInd   = (int*)(outf + O_IND);

    // prep: sort, gather caps, row map
    hipLaunchKernelGGL(k_prep, dim3(1), dim3(256), 0, stream,
                       lengths, caps_in, ind, dlen, caps, rmap, outCaps, outDlen, outInd);

    // weight conversions to bf16 (vectorized x4)
    auto conv = [&](const float* s, u16* d, int n) {
        const int n4 = n >> 2;
        hipLaunchKernelGGL(k_conv_bf16, dim3((n4 + 255) / 256), dim3(256), 0, stream,
                           s, (u32*)d, n4);
    };
    conv(W_enc_att, wenc,  DA * DENC);
    conv(W_dec_att, wdec,  DA * DH);
    conv(W_beta,    wbeta, DENC * DH);
    conv(W_ih,      wih,   DG * DX);
    conv(W_hh,      whh,   DG * DH);
    conv(W_fc,      wfc,   DV * DH);
    conv(W_init_h,  winh,  DH * DENC);
    conv(W_init_c,  winc,  DH * DENC);

    // mean over P (uses ind)
    hipLaunchKernelGGL(k_mean, dim3(DENC / 256, DB), dim3(256), 0, stream, enc_out, ind, mean);

    // h0 = mean @ W_init_h^T + b ; c0 likewise
    hipLaunchKernelGGL(k_gemm_bf16, dim3(DH / 64, 1), dim3(256), 0, stream,
                       mean, DB, DENC, winh, DH, b_init_h, h, 0, (const int*)nullptr);
    hipLaunchKernelGGL(k_gemm_bf16, dim3(DH / 64, 1), dim3(256), 0, stream,
                       mean, DB, DENC, winc, DH, b_init_c, c, 0, (const int*)nullptr);

    // enc_att = eo @ W_enc_att^T + b   (eo via row_map indirection, no 103MB copy)
    hipLaunchKernelGGL(k_gemm_bf16, dim3(DA / 64, MEO / 64), dim3(256), 0, stream,
                       enc_out, MEO, DENC, wenc, DA, b_enc_att, eatt, 0, rmap);

    // recurrent loop unrolled into the stream
    for (int t = 0; t < DT; ++t) {
        hipLaunchKernelGGL(k_gemm_bf16, dim3(DA / 64, 1), dim3(256), 0, stream,
                           h, DB, DH, wdec, DA, b_dec_att, dec, 0, (const int*)nullptr);
        hipLaunchKernelGGL(k_attn_scores, dim3(MEO / 8), dim3(256), 0, stream,
                           eatt, dec, W_full, b_full, scr);
        hipLaunchKernelGGL(k_softmax, dim3(DB), dim3(256), 0, stream,
                           scr, alph, outAlpha, dlen, t);
        hipLaunchKernelGGL(k_context, dim3(DENC / 256, DB), dim3(256), 0, stream,
                           enc_out, ind, alph, ctx);
        hipLaunchKernelGGL(k_gemm_bf16, dim3(DENC / 64, 1), dim3(256), 0, stream,
                           h, DB, DH, wbeta, DENC, b_beta, gate, 0, (const int*)nullptr);
        hipLaunchKernelGGL(k_build_x, dim3(DB * DX / 256), dim3(256), 0, stream,
                           emb, caps, t, gate, ctx, x);
        hipLaunchKernelGGL(k_gemm_bf16, dim3(DG / 64, 1), dim3(256), 0, stream,
                           x, DB, DX, wih, DG, b_ih, gts, 0, (const int*)nullptr);
        hipLaunchKernelGGL(k_gemm_bf16, dim3(DG / 64, 1), dim3(256), 0, stream,
                           h, DB, DH, whh, DG, b_hh, gts, 1, (const int*)nullptr);
        hipLaunchKernelGGL(k_lstm, dim3(DB * DH / 256), dim3(256), 0, stream,
                           gts, h, c, hn, dlen, t);
        hipLaunchKernelGGL(k_gemm_bf16, dim3((DV + 63) / 64, 1), dim3(256), 0, stream,
                           hn, DB, DH, wfc, DV, b_fc, prd, 0, (const int*)nullptr);
        hipLaunchKernelGGL(k_write_preds, dim3(DB * DV / 256), dim3(256), 0, stream,
                           prd, dlen, t, outPred);
    }
    (void)in_sizes; (void)n_in; (void)out_size; (void)ws_size;
}
